// SAGN_18734647345429
// MI455X (gfx1250) — matrix operand order repaired and checked by reading, not compiled, not run on hardware
//
#include <hip/hip_runtime.h>
#include <hip/hip_bf16.h>
#include <math.h>

// ---------------------------------------------------------------------------
// Problem constants (from the reference)
// ---------------------------------------------------------------------------
#define NN   8192          // nodes
#define HH   64            // hidden
#define NEG  139264        // edges (N*AVG_DEG + N self loops)
#define KK1  6554          // ceil(0.8*8192)
#define KK1P 6560          // KK1 padded to multiple of 16 (and of 32 for k-tiles)
#define KK2  5244          // ceil(0.8*6554)

typedef __attribute__((ext_vector_type(16))) _Float16 v16h;
typedef __attribute__((ext_vector_type(8)))  float    v8f;

#define DEV static __device__ __forceinline__

// ---------------------------------------------------------------------------
// Order-preserving float<->uint encoding for atomicMax on floats
// ---------------------------------------------------------------------------
DEV unsigned fenc(float f) {
    unsigned u = __float_as_uint(f);
    return (u & 0x80000000u) ? ~u : (u | 0x80000000u);
}
DEV float fdec(unsigned e) {
    unsigned u = (e & 0x80000000u) ? (e & 0x7fffffffu) : ~e;
    return __uint_as_float(u);
}
#define ENC_NEGINF 0x007FFFFFu   // fenc(-inf)

DEV float lrelu(float x) { return x > 0.f ? x : 0.2f * x; }
DEV float sigm(float x)  { return 1.f / (1.f + __expf(-x)); }

// ---------------------------------------------------------------------------
// Vector load helpers (aligned, unguarded -> global_load_b128)
// ---------------------------------------------------------------------------
DEV void ld8(const float* __restrict__ p, float* o) {
    const float4* q = (const float4*)p;
    float4 a = q[0], b = q[1];
    o[0] = a.x; o[1] = a.y; o[2] = a.z; o[3] = a.w;
    o[4] = b.x; o[5] = b.y; o[6] = b.z; o[7] = b.w;
}
// A-tile raw f32 load: raw[0..7]=A[m,k0+koff+0..7], raw[8..15]=A[m,k0+16+koff+0..7]
DEV void load_a_raw(const float* __restrict__ A, int lda, int m0, int k0, float* raw) {
    int lane = threadIdx.x & 31;
    int m    = m0 + (lane & 15);
    int koff = (lane >> 4) << 3;
    const float* base = A + (size_t)m * lda + k0 + koff;
    ld8(base,      raw);
    ld8(base + 16, raw + 8);
}
DEV v16h cvt16(const float* raw) {
    v16h a;
#pragma unroll
    for (int t = 0; t < 16; ++t) a[t] = (_Float16)raw[t];
    return a;
}
// B 32x16 f16 tile, unguarded (K dimension must be padded/in-range)
DEV v16h load_b16(const float* __restrict__ B, int ldb, int k0, int n0) {
    int lane = threadIdx.x & 31;
    int n  = n0 + (lane & 15);
    int kb = (lane >> 4) << 3;
    v16h b;
#pragma unroll
    for (int t = 0; t < 16; ++t)
        b[t] = (_Float16)B[(size_t)(k0 + ((t < 8) ? (kb + t) : (16 + kb + (t - 8)))) * ldb + n];
    return b;
}

// Unguarded store of one 16x16 D tile (destinations are padded; pad rows unread
// or re-zeroed by the host sequence).
template <int HAS_SCALE, int HAS_BIAS, int RELU>
DEV void store_d16(float* __restrict__ C, int ldc, int m0, int n0, v8f d,
                   const float* __restrict__ rowscale, const float* __restrict__ bias) {
    int lane = threadIdx.x & 31;
    int n  = n0 + (lane & 15);
    int mo = (lane >> 4) << 3;
    float bn = 0.f;
    if (HAS_BIAS) bn = bias[n];
#pragma unroll
    for (int v = 0; v < 8; ++v) {
        int m = m0 + mo + v;
        float x = d[v];
        if (HAS_SCALE) x *= rowscale[m];
        if (HAS_BIAS)  x += bn;
        if (RELU)      x = fmaxf(x, 0.f);
        C[(size_t)m * ldc + n] = x;
    }
}

// ---------------------------------------------------------------------------
// C[M,64] = A[M,64] @ B[64,64] (+bias), one 16x16 tile per wave.
// A and C must be readable/writable for ceil(M/16)*16 rows.
// ---------------------------------------------------------------------------
template <int HAS_BIAS>
__global__ void k_mm64_t(const float* __restrict__ A, int M,
                         const float* __restrict__ B,
                         const float* __restrict__ bias,
                         float* __restrict__ C) {
    int wid = blockIdx.x * (blockDim.x >> 5) + (threadIdx.x >> 5);
    int tilesM = (M + 15) >> 4;
    if (wid >= tilesM * 4) return;
    int m0 = (wid >> 2) * 16, n0 = (wid & 3) * 16;

    v8f acc;
#pragma unroll
    for (int i = 0; i < 8; ++i) acc[i] = 0.f;

#pragma unroll
    for (int k0 = 0; k0 < 64; k0 += 32) {
        float raw[16];
        load_a_raw(A, 64, m0, k0, raw);
        v16h a = cvt16(raw);
        v16h b = load_b16(B, 64, k0, n0);
        acc = __builtin_amdgcn_wmma_f32_16x16x32_f16(false, a, false, b, (short)0, acc, false, false);
    }
    store_d16<0, HAS_BIAS, 0>(C, 64, m0, n0, acc, nullptr, bias);
}

// ---------------------------------------------------------------------------
// C[M,64] = f(E2[M,K]) @ B[K,64], 16 rows x 64 cols per wave. lda = KK1P.
// E2 zero-padded [KK1P x KK1P]; B zero-padded to KK1P rows -> no load guards.
// MODE 0: f = identity (A_hat matmuls); MODE 1: f = masked-softmax alpha2.
// ---------------------------------------------------------------------------
template <int MODE, int HAS_SCALE, int HAS_BIAS, int RELU>
__global__ void k_mmE2_t(const float* __restrict__ A, int M, int K,
                         const float* __restrict__ B, float* __restrict__ C,
                         const float* __restrict__ rowscale, const float* __restrict__ bias,
                         const float* __restrict__ q2, const float* __restrict__ r2,
                         const float* __restrict__ gb_p,
                         const float* __restrict__ mrow, const float* __restrict__ invden) {
    int wid = blockIdx.x * (blockDim.x >> 5) + (threadIdx.x >> 5);
    int m0 = wid * 16;
    if (m0 >= M) return;
    int lane = threadIdx.x & 31;
    int m    = m0 + (lane & 15);
    int koff = (lane >> 4) << 3;

    float gb = 0.f, qm = 0.f, mm = 0.f, dd = 0.f;
    if (MODE) {
        gb = *gb_p;
        qm = q2[m]; mm = mrow[m]; dd = invden[m];
    }

    v8f acc[4];
#pragma unroll
    for (int t = 0; t < 4; ++t)
#pragma unroll
        for (int i = 0; i < 8; ++i) acc[t][i] = 0.f;

    for (int k0 = 0; k0 < K; k0 += 32) {
        float raw[16];
        load_a_raw(A, KK1P, m0, k0, raw);
        if (k0 + 96 < KK1P)
            __builtin_prefetch(&A[(size_t)m * KK1P + k0 + 96], 0, 0);  // global_prefetch_b8
        v16h a;
        if (MODE) {
#pragma unroll
            for (int t = 0; t < 16; ++t) {
                float e = raw[t];
                float v = 0.f;
                if (e > 0.f) {
                    int k = k0 + ((t < 8) ? (koff + t) : (16 + koff + (t - 8)));
                    float s = lrelu(qm + r2[k] + gb);
                    v = __expf(s - mm) * dd;
                }
                a[t] = (_Float16)v;
            }
        } else {
            a = cvt16(raw);
        }
#pragma unroll
        for (int tn = 0; tn < 4; ++tn) {
            v16h b = load_b16(B, 64, k0, tn * 16);
            acc[tn] = __builtin_amdgcn_wmma_f32_16x16x32_f16(false, a, false, b, (short)0, acc[tn], false, false);
        }
    }
#pragma unroll
    for (int tn = 0; tn < 4; ++tn)
        store_d16<HAS_SCALE, HAS_BIAS, RELU>(C, 64, m0, tn * 16, acc[tn], rowscale, bias);
}

// ---------------------------------------------------------------------------
// fills
// ---------------------------------------------------------------------------
__global__ void k_fill_f32(float* p, float v, size_t n) {
    for (size_t i = (size_t)blockIdx.x * blockDim.x + threadIdx.x; i < n; i += (size_t)gridDim.x * blockDim.x) p[i] = v;
}
__global__ void k_fill_u32(unsigned* p, unsigned v, size_t n) {
    for (size_t i = (size_t)blockIdx.x * blockDim.x + threadIdx.x; i < n; i += (size_t)gridDim.x * blockDim.x) p[i] = v;
}

// ---------------------------------------------------------------------------
// layer-1 edge-list kernels
// ---------------------------------------------------------------------------
__global__ void k_count_deg(const int* __restrict__ row, const int* __restrict__ col,
                            float* __restrict__ deg, float* __restrict__ deg1) {
    int e = blockIdx.x * blockDim.x + threadIdx.x;
    if (e >= NEG) return;
    int r = row[e];
    atomicAdd(&deg[r], 1.f);
    if (r != col[e]) atomicAdd(&deg1[r], 1.f);
}
__global__ void k_dinv(const float* __restrict__ deg, float* __restrict__ dinv) {
    int i = blockIdx.x * blockDim.x + threadIdx.x;
    if (i >= NN) return;
    float d = deg[i];
    dinv[i] = d > 0.f ? rsqrtf(d) : 0.f;
}
__global__ void k_fill_bias(float* __restrict__ out, const float* __restrict__ bias, int M) {
    int i = blockIdx.x * blockDim.x + threadIdx.x;
    if (i >= M * 64) return;
    out[i] = bias[i & 63];
}
__global__ void k_norm_scatter(const int* __restrict__ row, const int* __restrict__ col,
                               const float* __restrict__ dinv, const float* __restrict__ y,
                               float* __restrict__ out) {
    long long idx = (long long)blockIdx.x * blockDim.x + threadIdx.x;
    if (idx >= (long long)NEG * 64) return;
    int e = (int)(idx >> 6), h = (int)(idx & 63);
    int r = row[e], c = col[e];
    atomicAdd(&out[(size_t)r * 64 + h], dinv[r] * dinv[c] * y[(size_t)c * 64 + h]);
}
__global__ void k_relu(float* p, size_t n) {
    size_t i = (size_t)blockIdx.x * blockDim.x + threadIdx.x;
    if (i < n) p[i] = fmaxf(p[i], 0.f);
}
__global__ void k_segmax_scatter(const int* __restrict__ row, const int* __restrict__ col,
                                 const float* __restrict__ xp, unsigned* __restrict__ xqe) {
    long long idx = (long long)blockIdx.x * blockDim.x + threadIdx.x;
    if (idx >= (long long)NEG * 64) return;
    int e = (int)(idx >> 6), h = (int)(idx & 63);
    atomicMax(&xqe[(size_t)row[e] * 64 + h], fenc(xp[(size_t)col[e] * 64 + h]));
}
__global__ void k_dec(const unsigned* __restrict__ in, float* __restrict__ out, size_t n) {
    size_t i = (size_t)blockIdx.x * blockDim.x + threadIdx.x;
    if (i < n) out[i] = fdec(in[i]);
}
__global__ void k_dot64(const float* __restrict__ X, int M, const float* __restrict__ w,
                        const float* __restrict__ beta, float* __restrict__ out) {
    int i = blockIdx.x * blockDim.x + threadIdx.x;
    if (i >= M) return;
    float a = beta ? *beta : 0.f;
    const float* xr = X + (size_t)i * 64;
#pragma unroll 8
    for (int k = 0; k < 64; ++k) a += xr[k] * w[k];
    out[i] = a;
}
__global__ void k_edge_s(const int* __restrict__ row, const int* __restrict__ col,
                         const float* __restrict__ qa, const float* __restrict__ rb,
                         const float* __restrict__ gbp, float* __restrict__ s,
                         unsigned* __restrict__ menc) {
    int e = blockIdx.x * blockDim.x + threadIdx.x;
    if (e >= NEG) return;
    float v = lrelu(qa[row[e]] + rb[col[e]] + *gbp);
    s[e] = v;
    atomicMax(&menc[row[e]], fenc(v));
}
__global__ void k_edge_exp(const int* __restrict__ row, const float* __restrict__ s,
                           const unsigned* __restrict__ menc, float* __restrict__ t,
                           float* __restrict__ den) {
    int e = blockIdx.x * blockDim.x + threadIdx.x;
    if (e >= NEG) return;
    float tv = __expf(s[e] - fdec(menc[row[e]]));
    t[e] = tv;
    atomicAdd(&den[row[e]], tv);
}
__global__ void k_edge_alpha(const int* __restrict__ row, const float* __restrict__ t,
                             const float* __restrict__ den, float* __restrict__ alpha) {
    int e = blockIdx.x * blockDim.x + threadIdx.x;
    if (e >= NEG) return;
    alpha[e] = t[e] / den[row[e]];
}
__global__ void k_scatter_out1(const int* __restrict__ row, const int* __restrict__ col,
                               const float* __restrict__ alpha, const float* __restrict__ x0,
                               float* __restrict__ out1) {
    long long idx = (long long)blockIdx.x * blockDim.x + threadIdx.x;
    if (idx >= (long long)NEG * 64) return;
    int e = (int)(idx >> 6), h = (int)(idx & 63);
    atomicAdd(&out1[(size_t)row[e] * 64 + h], alpha[e] * x0[(size_t)col[e] * 64 + h]);
}
__global__ void k_aggr(const int* __restrict__ row, const int* __restrict__ col,
                       const float* __restrict__ sc, float* __restrict__ aggr) {
    int e = blockIdx.x * blockDim.x + threadIdx.x;
    if (e >= NEG) return;
    if (row[e] != col[e]) atomicAdd(&aggr[row[e]], sc[col[e]]);
}
__global__ void k_fit(const float* __restrict__ deg1, const float* __restrict__ t1,
                      const float* __restrict__ aggr, const float* __restrict__ t2,
                      float* __restrict__ fit) {
    int i = blockIdx.x * blockDim.x + threadIdx.x;
    if (i >= NN) return;
    fit[i] = sigm(deg1[i] * t1[i] + aggr[i] + t2[i]);
}

// ---------------------------------------------------------------------------
// single-workgroup bitonic sort of 8192 (value desc, index asc on ties)
// dynamic LDS: 8192*4 floats + 8192*4 ints = 64KB
// ---------------------------------------------------------------------------
__global__ void k_sort8192(const float* __restrict__ vin, float* __restrict__ vout, int* __restrict__ iout) {
    extern __shared__ char smem[];
    float* sv = (float*)smem;
    int*   si = (int*)(sv + 8192);
    const int n = 8192;
    for (int i = threadIdx.x; i < n; i += blockDim.x) { sv[i] = vin[i]; si[i] = i; }
    __syncthreads();
    for (int k = 2; k <= n; k <<= 1) {
        for (int j = k >> 1; j > 0; j >>= 1) {
            for (int i = threadIdx.x; i < n; i += blockDim.x) {
                int ixj = i ^ j;
                if (ixj > i) {
                    bool up = ((i & k) == 0);
                    float va = sv[i], vb = sv[ixj];
                    int   ia = si[i], ib = si[ixj];
                    bool before = (va > vb) || (va == vb && ia < ib);
                    bool sw = up ? !before : before;
                    if (sw) { sv[i] = vb; sv[ixj] = va; si[i] = ib; si[ixj] = ia; }
                }
            }
            __syncthreads();
        }
    }
    for (int i = threadIdx.x; i < n; i += blockDim.x) { vout[i] = sv[i]; iout[i] = si[i]; }
}

__global__ void k_gather_scale(float* __restrict__ out, const float* __restrict__ src,
                               const int* __restrict__ perm, const float* __restrict__ vals,
                               int k, int pad) {
    int idx = blockIdx.x * blockDim.x + threadIdx.x;
    if (idx >= pad * 64) return;
    int a = idx >> 6, h = idx & 63;
    out[idx] = (a < k) ? src[(size_t)perm[a] * 64 + h] * vals[a] : 0.f;
}
__global__ void k_setperm(const int* __restrict__ perm, int* __restrict__ nidx, int* __restrict__ inp) {
    int a = blockIdx.x * blockDim.x + threadIdx.x;
    if (a >= KK1) return;
    int node = perm[a];
    nidx[node] = a;
    inp[node]  = 1;
}

// ---------------------------------------------------------------------------
// CSR over col[] (sparse S rows) and E2 = S^T A S accumulation (ld = KK1P)
// ---------------------------------------------------------------------------
__global__ void k_count_col(const int* __restrict__ col, int* __restrict__ cnt) {
    int e = blockIdx.x * blockDim.x + threadIdx.x;
    if (e < NEG) atomicAdd(&cnt[col[e]], 1);
}
__global__ void k_scan_csr(const int* __restrict__ cnt, int* __restrict__ off, int* __restrict__ cur) {
    __shared__ int part[1024];
    int t = threadIdx.x, base = t * 8;
    int s = 0;
#pragma unroll
    for (int u = 0; u < 8; ++u) s += cnt[base + u];
    part[t] = s;
    __syncthreads();
    if (t == 0) {
        int run = 0;
        for (int i = 0; i < 1024; ++i) { int tmp = part[i]; part[i] = run; run += tmp; }
        off[NN] = run;
    }
    __syncthreads();
    int run = part[t];
#pragma unroll
    for (int u = 0; u < 8; ++u) {
        off[base + u] = run;
        cur[base + u] = run;
        run += cnt[base + u];
    }
}
__global__ void k_fill_entries(const int* __restrict__ row, const int* __restrict__ col,
                               const int* __restrict__ nidx, const int* __restrict__ inp,
                               const float* __restrict__ alpha,
                               int* __restrict__ cur, int* __restrict__ enta, float* __restrict__ entv) {
    int e = blockIdx.x * blockDim.x + threadIdx.x;
    if (e >= NEG) return;
    int p = atomicAdd(&cur[col[e]], 1);
    int rr = row[e];
    int in = inp[rr];
    enta[p] = in ? nidx[rr] : 0;
    entv[p] = in ? alpha[e] : 0.f;
}
__global__ void k_E2_accum(const int* __restrict__ row, const int* __restrict__ col,
                           const int* __restrict__ off, const int* __restrict__ enta,
                           const float* __restrict__ entv, float* __restrict__ E2) {
    int e = blockIdx.x * blockDim.x + threadIdx.x;
    if (e >= NEG) return;
    int i = row[e], j = col[e];
    int s0 = off[i], e0 = off[i + 1];
    int s1 = off[j], e1 = off[j + 1];
    for (int u = s0; u < e0; ++u) {
        float va = entv[u];
        if (va == 0.f) continue;
        size_t base = (size_t)enta[u] * KK1P;
        for (int w = s1; w < e1; ++w) {
            float vb = entv[w];
            if (vb != 0.f) atomicAdd(&E2[base + enta[w]], va * vb);
        }
    }
}
__global__ void k_diag1(float* __restrict__ E2) {
    int a = blockIdx.x * blockDim.x + threadIdx.x;
    if (a < KK1) E2[(size_t)a * KK1P + a] = 1.0f;
}

// ---------------------------------------------------------------------------
// coarse-graph row kernels (one wave per row, wave32 shuffles)
// ---------------------------------------------------------------------------
__global__ void k_rowsum(const float* __restrict__ E2, float* __restrict__ degd, float* __restrict__ dinv) {
    int r = blockIdx.x * 8 + (threadIdx.x >> 5);
    if (r >= KK1) return;
    int lane = threadIdx.x & 31;
    float s = 0.f;
    const float* er = E2 + (size_t)r * KK1P;
    for (int j = lane; j < KK1; j += 32) s += er[j];
#pragma unroll
    for (int o = 16; o; o >>= 1) s += __shfl_xor(s, o, 32);
    if (lane == 0) { degd[r] = s; dinv[r] = s > 0.f ? rsqrtf(s) : 0.f; }
}
__global__ void k_colmax(const float* __restrict__ E2, const float* __restrict__ xp2,
                         float* __restrict__ xq2) {
    int r = blockIdx.x * 8 + (threadIdx.x >> 5);
    if (r >= KK1) return;
    int lane = threadIdx.x & 31;
    float v0 = -INFINITY, v1 = -INFINITY;
    const float* er = E2 + (size_t)r * KK1P;
    for (int j = 0; j < KK1; ++j) {
        if (er[j] > 0.f) {
            v0 = fmaxf(v0, xp2[(size_t)j * 64 + lane]);
            v1 = fmaxf(v1, xp2[(size_t)j * 64 + lane + 32]);
        }
    }
    xq2[(size_t)r * 64 + lane]      = v0;
    xq2[(size_t)r * 64 + lane + 32] = v1;
}
__global__ void k_stats(const float* __restrict__ E2, const float* __restrict__ q2,
                        const float* __restrict__ r2, const float* __restrict__ gbp,
                        float* __restrict__ m2, float* __restrict__ invden) {
    int r = blockIdx.x * 8 + (threadIdx.x >> 5);
    if (r >= KK1) return;
    int lane = threadIdx.x & 31;
    float gb = *gbp, q = q2[r];
    float m = -INFINITY, s = 0.f;
    const float* er = E2 + (size_t)r * KK1P;
    for (int j = lane; j < KK1; j += 32) {
        if (er[j] > 0.f) {
            float v = lrelu(q + r2[j] + gb);
            if (v > m) { s = s * __expf(m - v) + 1.f; m = v; }
            else       { s += __expf(v - m); }
        }
    }
#pragma unroll
    for (int o = 16; o; o >>= 1) {
        float mo = __shfl_xor(m, o, 32);
        float so = __shfl_xor(s, o, 32);
        float mn = fmaxf(m, mo);
        float sn = (s  > 0.f ? s  * __expf(m  - mn) : 0.f) +
                   (so > 0.f ? so * __expf(mo - mn) : 0.f);
        m = mn; s = sn;
    }
    if (lane == 0) { m2[r] = m; invden[r] = 1.f / s; }
}
__global__ void k_matvec2(const float* __restrict__ E2, const float* __restrict__ sc2,
                          float* __restrict__ aggr2) {
    int r = blockIdx.x * 8 + (threadIdx.x >> 5);
    if (r >= KK1) return;
    int lane = threadIdx.x & 31;
    float s = 0.f;
    const float* er = E2 + (size_t)r * KK1P;
    for (int j = lane; j < KK1; j += 32) s += er[j] * sc2[j];
#pragma unroll
    for (int o = 16; o; o >>= 1) s += __shfl_xor(s, o, 32);
    if (lane == 0) aggr2[r] = s - sc2[r];   // Wns has zero diagonal; E2 diag == 1
}
__global__ void k_fit2(const float* __restrict__ degd, const float* __restrict__ t1b,
                       const float* __restrict__ aggr2, const float* __restrict__ t2b,
                       float* __restrict__ fit2p) {
    int i = blockIdx.x * blockDim.x + threadIdx.x;
    if (i >= 8192) return;
    fit2p[i] = (i < KK1) ? sigm((degd[i] - 1.f) * t1b[i] + aggr2[i] + t2b[i]) : -INFINITY;
}
__global__ void k_scale_rows(float* __restrict__ dst, const float* __restrict__ src,
                             const float* __restrict__ scale, int M) {
    int idx = blockIdx.x * blockDim.x + threadIdx.x;
    if (idx >= M * 64) return;
    dst[idx] = src[idx] * scale[idx >> 6];
}

// ---------------------------------------------------------------------------
// readout + MLP head
// ---------------------------------------------------------------------------
__global__ void k_colstat(const float* __restrict__ X, int M, float* __restrict__ xs, int add) {
    __shared__ float ssum[256];
    __shared__ float smax[256];
    int h = blockIdx.x, t = threadIdx.x;
    float s = 0.f, mx = -INFINITY;
    for (int i = t; i < M; i += blockDim.x) {
        float v = X[(size_t)i * 64 + h];
        s += v; mx = fmaxf(mx, v);
    }
    ssum[t] = s; smax[t] = mx;
    __syncthreads();
    for (int o = 128; o > 0; o >>= 1) {
        if (t < o) { ssum[t] += ssum[t + o]; smax[t] = fmaxf(smax[t], smax[t + o]); }
        __syncthreads();
    }
    if (t == 0) {
        float mean = ssum[0] / (float)M;
        if (add) { xs[h] += mean; xs[64 + h] += smax[0]; }
        else     { xs[h]  = mean; xs[64 + h]  = smax[0]; }
    }
}
__global__ void k_final(const float* __restrict__ xs,
                        const float* __restrict__ W1, const float* __restrict__ b1,
                        const float* __restrict__ W2, const float* __restrict__ b2,
                        float* __restrict__ out) {
    __shared__ float z1[64];
    __shared__ float z2[64];
    __shared__ float red[3];
    int t = threadIdx.x;
    if (t < 64) {
        float a = b1[t];
        for (int k = 0; k < 128; ++k) a += xs[k] * W1[k * 64 + t];
        z1[t] = fmaxf(a, 0.f);
    }
    __syncthreads();
    if (t < 64) {
        float a = b2[t];
        for (int k = 0; k < 64; ++k) a += z1[k] * W2[k * 64 + t];
        z2[t] = a;
    }
    __syncthreads();
    if (t == 0) {
        float m = -INFINITY;
        for (int i = 0; i < 64; ++i) m = fmaxf(m, z2[i]);
        float s = 0.f;
        for (int i = 0; i < 64; ++i) s += __expf(z2[i] - m);
        red[0] = m + __logf(s);
    }
    __syncthreads();
    float lse = red[0];
    if (t < 64) z2[t] = z2[t] - lse;     // log_softmax
    __syncthreads();
    if (t == 0) {
        float mu = 0.f;
        for (int i = 0; i < 64; ++i) mu += z2[i];
        mu /= 64.f;
        float var = 0.f;
        for (int i = 0; i < 64; ++i) { float d = z2[i] - mu; var += d * d; }
        var /= 64.f;
        red[1] = mu; red[2] = rsqrtf(var + 1e-5f);
    }
    __syncthreads();
    if (t < 64) out[t] = (z2[t] - red[1]) * red[2];
}

// ---------------------------------------------------------------------------
// host launcher (requires ~210 MB workspace; E2 [6560^2] = 172 MB fits in L2)
// ---------------------------------------------------------------------------
extern "C" void kernel_launch(void* const* d_in, const int* in_sizes, int n_in,
                              void* d_out, int out_size, void* d_ws, size_t ws_size,
                              hipStream_t stream) {
    (void)in_sizes; (void)n_in; (void)out_size; (void)ws_size;
    const float* x    = (const float*)d_in[0];
    const int*   row  = (const int*)d_in[1];
    const int*   col  = row + NEG;
    const float* conv1_w = (const float*)d_in[3],  *conv1_b = (const float*)d_in[4];
    const float* p1_iw = (const float*)d_in[5],    *p1_ib   = (const float*)d_in[6];
    const float* p1_qw = (const float*)d_in[7],    *p1_qb   = (const float*)d_in[8];
    const float* p1_gw = (const float*)d_in[9],    *p1_gb   = (const float*)d_in[10];
    const float* p1_scw  = (const float*)d_in[11];
    const float* p1_l1w  = (const float*)d_in[12], *p1_l1b  = (const float*)d_in[13];
    const float* p1_l2w  = (const float*)d_in[14], *p1_l2b  = (const float*)d_in[15];
    const float* conv2_w = (const float*)d_in[16], *conv2_b = (const float*)d_in[17];
    const float* p2_iw = (const float*)d_in[18],   *p2_ib   = (const float*)d_in[19];
    const float* p2_qw = (const float*)d_in[20],   *p2_qb   = (const float*)d_in[21];
    const float* p2_gw = (const float*)d_in[22],   *p2_gb   = (const float*)d_in[23];
    const float* p2_scw  = (const float*)d_in[24];
    const float* p2_l1w  = (const float*)d_in[25], *p2_l1b  = (const float*)d_in[26];
    const float* p2_l2w  = (const float*)d_in[27], *p2_l2b  = (const float*)d_in[28];
    const float* lin1_w  = (const float*)d_in[29], *lin1_b  = (const float*)d_in[30];
    const float* lin2_w  = (const float*)d_in[31], *lin2_b  = (const float*)d_in[32];

    // bump allocator over d_ws
    char* wsb = (char*)d_ws;
    size_t curo = 0;
    auto alloc = [&](size_t bytes) -> char* {
        char* r = wsb + curo;
        curo = (curo + bytes + 255) & ~(size_t)255;
        return r;
    };
    float* deg   = (float*)alloc(NN * 4);
    float* deg1  = (float*)alloc(NN * 4);
    float* dinv  = (float*)alloc(NN * 4);
    float* ybuf  = (float*)alloc((size_t)NN * 64 * 4);
    float* x0    = (float*)alloc((size_t)NN * 64 * 4);
    float* xp    = (float*)alloc((size_t)NN * 64 * 4);
    unsigned* xqe = (unsigned*)alloc((size_t)NN * 64 * 4);
    float* xq    = (float*)alloc((size_t)NN * 64 * 4);
    float* mq    = (float*)alloc((size_t)NN * 64 * 4);
    float* qa    = (float*)alloc(NN * 4);
    float* rb    = (float*)alloc(NN * 4);
    float* sbuf  = (float*)alloc((size_t)NEG * 4);
    float* tbuf  = (float*)alloc((size_t)NEG * 4);
    float* alpha = (float*)alloc((size_t)NEG * 4);
    unsigned* menc = (unsigned*)alloc(NN * 4);
    float* den   = (float*)alloc(NN * 4);
    float* out1  = (float*)alloc((size_t)NN * 64 * 4);
    float* sc    = (float*)alloc(NN * 4);
    float* t1    = (float*)alloc(NN * 4);
    float* t2    = (float*)alloc(NN * 4);
    float* aggr  = (float*)alloc(NN * 4);
    float* fit   = (float*)alloc(NN * 4);
    float* sortv = (float*)alloc(8192 * 4);
    int*   sorti = (int*)alloc(8192 * 4);
    float* x1    = (float*)alloc((size_t)KK1P * 64 * 4);
    int*   nidx  = (int*)alloc(NN * 4);
    int*   inp   = (int*)alloc(NN * 4);
    int*   cnt   = (int*)alloc(NN * 4);
    int*   offs  = (int*)alloc((NN + 1) * 4);
    int*   curp  = (int*)alloc(NN * 4);
    int*   enta  = (int*)alloc((size_t)NEG * 4);
    float* entv  = (float*)alloc((size_t)NEG * 4);
    float* E2    = (float*)alloc((size_t)KK1P * KK1P * 4);
    float* degd  = (float*)alloc(KK1P * 4);
    float* dinvd = (float*)alloc(KK1P * 4);
    float* y2    = (float*)alloc((size_t)KK1P * 64 * 4);
    float* y2s   = (float*)alloc((size_t)KK1P * 64 * 4);
    float* x2    = (float*)alloc((size_t)KK1P * 64 * 4);
    float* xp2   = (float*)alloc((size_t)KK1P * 64 * 4);
    float* xq2   = (float*)alloc((size_t)KK1P * 64 * 4);
    float* mq2   = (float*)alloc((size_t)KK1P * 64 * 4);
    float* q2    = (float*)alloc(KK1P * 4);
    float* r2    = (float*)alloc(KK1P * 4);
    float* m2    = (float*)alloc(KK1P * 4);
    float* invden = (float*)alloc(KK1P * 4);
    float* out2  = (float*)alloc((size_t)KK1P * 64 * 4);
    float* sc2   = (float*)alloc(KK1P * 4);
    float* t1b   = (float*)alloc(KK1P * 4);
    float* t2b   = (float*)alloc(KK1P * 4);
    float* aggr2 = (float*)alloc(KK1P * 4);
    float* fit2p = (float*)alloc(8192 * 4);
    float* sv2   = (float*)alloc(8192 * 4);
    int*   si2   = (int*)alloc(8192 * 4);
    float* x3    = (float*)alloc((size_t)KK2 * 64 * 4);
    float* xs    = (float*)alloc(128 * 4);

    const int T = 256;
    const int gN   = (NN + T - 1) / T;
    const int gE   = (NEG + T - 1) / T;
    const int gEH  = (int)(((long long)NEG * 64 + T - 1) / T);
    const int gNH  = (int)(((long long)NN * 64 + T - 1) / T);
    const int gK1  = (KK1 + T - 1) / T;
    const int gK1H = (KK1P * 64 + T - 1) / T;
    const int gRow = (KK1 + 7) / 8;          // wave-per-row kernels, 8 waves/block
    auto mm64_blocks = [](int M) { return ((((M + 15) >> 4) * 4) + 7) / 8; };
    const int mmE2_blocks = (((KK1 + 15) >> 4) + 7) / 8;

    // ---- degrees ----
    k_fill_f32<<<64, T, 0, stream>>>(deg, 0.f, NN);
    k_fill_f32<<<64, T, 0, stream>>>(deg1, 0.f, NN);
    k_count_deg<<<gE, T, 0, stream>>>(row, col, deg, deg1);
    k_dinv<<<gN, T, 0, stream>>>(deg, dinv);

    // ---- conv1 + relu ----
    k_mm64_t<0><<<mm64_blocks(NN), T, 0, stream>>>(x, NN, conv1_w, nullptr, ybuf);
    k_fill_bias<<<gNH, T, 0, stream>>>(x0, conv1_b, NN);
    k_norm_scatter<<<gEH, T, 0, stream>>>(row, col, dinv, ybuf, x0);
    k_relu<<<gNH, T, 0, stream>>>(x0, (size_t)NN * 64);

    // ---- ASAP pool 1: xp = gcn_sparse(x0) ----
    k_mm64_t<0><<<mm64_blocks(NN), T, 0, stream>>>(x0, NN, p1_iw, nullptr, ybuf);
    k_fill_bias<<<gNH, T, 0, stream>>>(xp, p1_ib, NN);
    k_norm_scatter<<<gEH, T, 0, stream>>>(row, col, dinv, ybuf, xp);

    // ---- Xq = seg_max(xp[col], row); Mq ----
    k_fill_u32<<<512, T, 0, stream>>>(xqe, ENC_NEGINF, (size_t)NN * 64);
    k_segmax_scatter<<<gEH, T, 0, stream>>>(row, col, xp, xqe);
    k_dec<<<gNH, T, 0, stream>>>(xqe, xq, (size_t)NN * 64);
    k_mm64_t<1><<<mm64_blocks(NN), T, 0, stream>>>(xq, NN, p1_qw, p1_qb, mq);

    // ---- edge attention softmax ----
    k_dot64<<<gN, T, 0, stream>>>(mq, NN, p1_gw, nullptr, qa);
    k_dot64<<<gN, T, 0, stream>>>(xp, NN, p1_gw + 64, nullptr, rb);
    k_fill_u32<<<64, T, 0, stream>>>(menc, ENC_NEGINF, NN);
    k_edge_s<<<gE, T, 0, stream>>>(row, col, qa, rb, p1_gb, sbuf, menc);
    k_fill_f32<<<64, T, 0, stream>>>(den, 0.f, NN);
    k_edge_exp<<<gE, T, 0, stream>>>(row, sbuf, menc, tbuf, den);
    k_edge_alpha<<<gE, T, 0, stream>>>(row, tbuf, den, alpha);
    k_fill_f32<<<512, T, 0, stream>>>(out1, 0.f, (size_t)NN * 64);
    k_scatter_out1<<<gEH, T, 0, stream>>>(row, col, alpha, x0, out1);

    // ---- LEConv fitness + top-k1 ----
    k_dot64<<<gN, T, 0, stream>>>(out1, NN, p1_scw, nullptr, sc);
    k_dot64<<<gN, T, 0, stream>>>(out1, NN, p1_l1w, p1_l1b, t1);
    k_dot64<<<gN, T, 0, stream>>>(out1, NN, p1_l2w, p1_l2b, t2);
    k_fill_f32<<<64, T, 0, stream>>>(aggr, 0.f, NN);
    k_aggr<<<gE, T, 0, stream>>>(row, col, sc, aggr);
    k_fit<<<gN, T, 0, stream>>>(deg1, t1, aggr, t2, fit);
    k_sort8192<<<1, 1024, 65536, stream>>>(fit, sortv, sorti);
    k_gather_scale<<<gK1H, T, 0, stream>>>(x1, out1, sorti, sortv, KK1, KK1P);

    // ---- E2 = S^T A S via CSR of sparse S rows ----
    k_fill_u32<<<64, T, 0, stream>>>((unsigned*)nidx, 0u, NN);
    k_fill_u32<<<64, T, 0, stream>>>((unsigned*)inp, 0u, NN);
    k_setperm<<<gK1, T, 0, stream>>>(sorti, nidx, inp);
    k_fill_u32<<<64, T, 0, stream>>>((unsigned*)cnt, 0u, NN);
    k_count_col<<<gE, T, 0, stream>>>(col, cnt);
    k_scan_csr<<<1, 1024, 0, stream>>>(cnt, offs, curp);
    k_fill_entries<<<gE, T, 0, stream>>>(row, col, nidx, inp, alpha, curp, enta, entv);
    k_fill_f32<<<4096, T, 0, stream>>>(E2, 0.f, (size_t)KK1P * KK1P);
    k_E2_accum<<<gE, T, 0, stream>>>(row, col, offs, enta, entv, E2);
    k_diag1<<<gK1, T, 0, stream>>>(E2);

    // ---- zero B-side operands of the WMMA matmuls (pad rows must be 0) ----
    k_fill_f32<<<64, T, 0, stream>>>(y2s, 0.f, (size_t)KK1P * 64);
    k_fill_f32<<<64, T, 0, stream>>>(x2, 0.f, (size_t)KK1P * 64);

    // ---- coarse layer: x2 = relu(A_hat @ (x1@W2) + b2) ----
    k_rowsum<<<gRow, T, 0, stream>>>(E2, degd, dinvd);
    k_mm64_t<0><<<mm64_blocks(KK1), T, 0, stream>>>(x1, KK1, conv2_w, nullptr, y2);
    k_scale_rows<<<gK1H, T, 0, stream>>>(y2s, y2, dinvd, KK1);
    k_mmE2_t<0, 1, 1, 1><<<mmE2_blocks, T, 0, stream>>>(E2, KK1, KK1, y2s, x2, dinvd, conv2_b,
                                                        nullptr, nullptr, nullptr, nullptr, nullptr);
    // re-zero x2 pad rows (stores above are unguarded; pads feed B-side later)
    k_fill_f32<<<1, T, 0, stream>>>(x2 + (size_t)KK1 * 64, 0.f, (size_t)(KK1P - KK1) * 64);

    // xp2 = A_hat @ (x2@Wi) + bi
    k_mm64_t<0><<<mm64_blocks(KK1), T, 0, stream>>>(x2, KK1, p2_iw, nullptr, y2);
    k_scale_rows<<<gK1H, T, 0, stream>>>(y2s, y2, dinvd, KK1);
    k_mmE2_t<0, 1, 1, 0><<<mmE2_blocks, T, 0, stream>>>(E2, KK1, KK1, y2s, xp2, dinvd, p2_ib,
                                                        nullptr, nullptr, nullptr, nullptr, nullptr);

    // ---- masked colmax, Mq2, fused masked-softmax attention out2 ----
    k_colmax<<<gRow, T, 0, stream>>>(E2, xp2, xq2);
    k_mm64_t<1><<<mm64_blocks(KK1), T, 0, stream>>>(xq2, KK1, p2_qw, p2_qb, mq2);
    k_dot64<<<gK1, T, 0, stream>>>(mq2, KK1, p2_gw, nullptr, q2);
    k_dot64<<<gK1, T, 0, stream>>>(xp2, KK1, p2_gw + 64, nullptr, r2);
    k_stats<<<gRow, T, 0, stream>>>(E2, q2, r2, p2_gb, m2, invden);
    k_mmE2_t<1, 0, 0, 0><<<mmE2_blocks, T, 0, stream>>>(E2, KK1, KK1, x2, out2, nullptr, nullptr,
                                                        q2, r2, p2_gb, m2, invden);

    // ---- layer-2 LEConv fitness + top-k2 ----
    k_dot64<<<gK1, T, 0, stream>>>(out2, KK1, p2_scw, nullptr, sc2);
    k_dot64<<<gK1, T, 0, stream>>>(out2, KK1, p2_l1w, p2_l1b, t1b);
    k_dot64<<<gK1, T, 0, stream>>>(out2, KK1, p2_l2w, p2_l2b, t2b);
    k_matvec2<<<gRow, T, 0, stream>>>(E2, sc2, aggr2);
    k_fit2<<<32, T, 0, stream>>>(degd, t1b, aggr2, t2b, fit2p);
    k_sort8192<<<1, 1024, 65536, stream>>>(fit2p, sv2, si2);
    k_gather_scale<<<(KK2 * 64 + T - 1) / T, T, 0, stream>>>(x3, out2, si2, sv2, KK2, KK2);

    // ---- readout + head ----
    k_colstat<<<64, 256, 0, stream>>>(x1, KK1, xs, 0);
    k_colstat<<<64, 256, 0, stream>>>(x3, KK2, xs, 1);
    k_final<<<1, 64, 0, stream>>>(xs, lin1_w, lin1_b, lin2_w, lin2_b, (float*)d_out);
}
